// MS_MoE_Conv_Temporal_7301444403350
// MI455X (gfx1250) — matrix operand order repaired and checked by reading, not compiled, not run on hardware
//
#include <hip/hip_runtime.h>
#include <hip/hip_bf16.h>
#include <math.h>

typedef __attribute__((ext_vector_type(16))) _Float16 v16h;
typedef __attribute__((ext_vector_type(8)))  _Float16 v8h;
typedef __attribute__((ext_vector_type(8)))  float    v8f;

#define T_ 4
#define B_ 16
#define C_ 384
#define HW_ 256
#define E_ 8
#define KT_ 12                /* C_/32 K-tiles */
#define NSLICE (B_*C_*HW_)    /* 1,572,864 elements per time step */
#define GEMM_LDS (4 * KT_ * 2 * 512 * sizeof(_Float16))   /* 96 KB */

// ---------------------------------------------------------------- utilities
__global__ void zero_f32(float* __restrict__ p, int n) {
    int i = blockIdx.x * 256 + threadIdx.x;
    if (i < n) p[i] = 0.f;
}

__global__ void cvt_f32_f16(const float* __restrict__ s, _Float16* __restrict__ d, int n) {
    int i = blockIdx.x * 256 + threadIdx.x;
    if (i < n) d[i] = (_Float16)s[i];
}

// LIF over T=4 steps: v += (x - v)/tau; spike = (v>=1); hard reset (detached).
// Spikes are exactly {0,1} -> stored as f16 for the WMMA GEMM.
__global__ void lif_spike(const float* __restrict__ src, _Float16* __restrict__ spk,
                          float invTau) {
    int j = blockIdx.x * 256 + threadIdx.x;    // grid sized exactly to NSLICE
    float v = 0.f;
    #pragma unroll
    for (int t = 0; t < T_; ++t) {
        float xv = src[(size_t)t * NSLICE + j];
        v += (xv - v) * invTau;
        float s = (v >= 1.f) ? 1.f : 0.f;
        spk[(size_t)t * NSLICE + j] = (_Float16)s;
        v = (s > 0.f) ? 0.f : v;
    }
}

// ------------------------------------------------------------------ router
// Fused LIF(tau=2) + 384->8 conv + sum over (t,c,hw). Deterministic partial
// sums per (b, slice) -- no float atomics so the output is bitwise stable.
__global__ void router_partial(const float* __restrict__ x,
                               const float* __restrict__ rW,
                               float* __restrict__ partial) {
    int b = blockIdx.x;
    int slice = blockIdx.y;                       // 8 slices of (c,hw) space
    __shared__ float sW[E_ * C_];                 // 12 KB router weights
    __shared__ float red[256];
    for (int i = threadIdx.x; i < E_ * C_; i += 256) sW[i] = rW[i];
    __syncthreads();

    float acc[E_] = {};
    for (int j = 0; j < 48; ++j) {                // 8*48*256 = 98304 = C_*HW_
        int p = slice * 12288 + j * 256 + threadIdx.x;
        int c = p >> 8, hw = p & 255;
        const float* xp = x + ((size_t)b * C_ + c) * HW_ + hw;
        float v = 0.f, ssum = 0.f;
        #pragma unroll
        for (int t = 0; t < T_; ++t) {
            float xv = xp[(size_t)t * NSLICE];
            v += (xv - v) * 0.5f;                 // tau = 2.0
            float s = (v >= 1.f) ? 1.f : 0.f;
            ssum += s;
            v = (s > 0.f) ? 0.f : v;
        }
        #pragma unroll
        for (int e = 0; e < E_; ++e) acc[e] += ssum * sW[e * C_ + c];
    }
    #pragma unroll
    for (int e = 0; e < E_; ++e) {
        red[threadIdx.x] = acc[e];
        __syncthreads();
        for (int s = 128; s > 0; s >>= 1) {
            if (threadIdx.x < s) red[threadIdx.x] += red[threadIdx.x + s];
            __syncthreads();
        }
        if (threadIdx.x == 0) partial[(b * 8 + slice) * E_ + e] = red[0];
        __syncthreads();
    }
}

// Fold BN affine, softmax over 8 experts, top-2, normalized gate weights.
__global__ void router_topk(const float* __restrict__ partial,
                            const float* __restrict__ rb,
                            const float* __restrict__ rs,
                            const float* __restrict__ rbb,
                            float* __restrict__ dense) {
    int b = threadIdx.x;
    if (b >= B_) return;
    const float invs = 1.f / sqrtf(1.f + 1e-5f);
    float lg[E_];
    #pragma unroll
    for (int e = 0; e < E_; ++e) {
        float s = 0.f;
        for (int sl = 0; sl < 8; ++sl) s += partial[(b * 8 + sl) * E_ + e];
        float mean = s * (1.f / (float)(T_ * HW_));
        lg[e] = (mean + rb[e]) * (rs[e] * invs) + rbb[e];
    }
    float mx = lg[0];
    for (int e = 1; e < E_; ++e) mx = fmaxf(mx, lg[e]);
    float pr[E_], sum = 0.f;
    for (int e = 0; e < E_; ++e) { pr[e] = expf(lg[e] - mx); sum += pr[e]; }
    for (int e = 0; e < E_; ++e) pr[e] /= sum;
    int i0 = 0;
    for (int e = 1; e < E_; ++e) if (pr[e] > pr[i0]) i0 = e;
    int i1 = (i0 == 0) ? 1 : 0;
    for (int e = 0; e < E_; ++e) if (e != i0 && pr[e] > pr[i1]) i1 = e;
    float w = pr[i0] + pr[i1];
    for (int e = 0; e < E_; ++e) dense[b * E_ + e] = 0.f;
    dense[b * E_ + i0] = pr[i0] / w;
    dense[b * E_ + i1] = pr[i1] / w;
}

// ------------------------------------------------- time-fused WMMA GEMM
// Block = 8 waves = 128(M) x 32(N) x ALL 4 time steps. kt-outer/t-inner:
// one A (weight) fragment feeds 8 WMMAs. Spike fragments for all 4 t live
// in 96 KB of dynamic LDS (CDNA5 WGP has 320 KB), staged once, one barrier.
//   mode 0: h = resid + BN(D + bias) is computed for all 4 t in registers,
//           stored, AND run through the LIF recursion -> spkOut (fuses the
//           second lif_spike pass into the GEMM epilogue).
//   mode 1: dst += gate * (BN(D + bias) + resid)   (final accumulate)
union AFrag  { v16h v; v8h h[2]; };
union V8Cvt  { v8h v; _Float16 h[8]; };

__global__ void expert_gemm_t4(const _Float16* __restrict__ S,     // spikes (T,B,C,HW) f16
                               const _Float16* __restrict__ Wh,    // weights [C][C] f16 (expert slice)
                               const float* __restrict__ bias,     // [C] (expert slice)
                               const float* __restrict__ bnS,
                               const float* __restrict__ bnB,
                               const float* __restrict__ resid,    // (T,B,C,HW) f32
                               float* __restrict__ dst,            // h (mode 0) or out (mode 1)
                               _Float16* __restrict__ spkOut,      // stage-2 spikes (mode 0)
                               const float* __restrict__ dense_w,  // (B,E) gates
                               float invTau, int expert, int mode) {
    extern __shared__ char smem_raw[];
    _Float16* sB = (_Float16*)smem_raw;        // [t][kt][nt][512], fragment order

    int nb   = blockIdx.x;                     // 0..127 : b*8 + hw-slab
    int b    = nb >> 3;
    int hw0  = (nb & 7) << 5;                  // 32-wide N slab
    int wave = threadIdx.x >> 5;               // 8 waves -> 8 M-tiles of 16
    int lane = threadIdx.x & 31;
    int m0   = blockIdx.y * 128 + wave * 16;

    // ---- stage spike tiles for ALL 4 time steps (fragment-ordered)
    for (int t = 0; t < T_; ++t) {
        const _Float16* Sb = S + (size_t)(t * B_ + b) * C_ * HW_ + hw0;
        #pragma unroll
        for (int i = 0; i < 6; ++i) {
            int g   = i * 256 + threadIdx.x;   // 1536 octets = 384x32 halves
            int kt  = g >> 7;
            int rem = g & 127;
            int lk  = rem >> 2;                // K row in tile (0..31)
            int ln8 = (rem & 3) * 8;           // N octet (0,8,16,24)
            V8Cvt cv;
            cv.v = *(const v8h*)(Sb + (size_t)(kt * 32 + lk) * HW_ + ln8);
            int nt8 = ln8 >> 4, nn0 = ln8 & 15;
            int klo = (lk & 7) | ((lk & 16) >> 1), khi = (lk & 8) << 1;
            _Float16* basep = sB + ((t * KT_ + kt) * 2 + nt8) * 512 + khi * 16 + klo;
            #pragma unroll
            for (int j = 0; j < 8; ++j)
                basep[(nn0 + j) * 16] = cv.h[j];   // frag lane = n + khi, elem = klo
        }
    }
    __syncthreads();                           // the ONLY barrier

    // ---- kt-outer, t/nt-inner: A fragment reused by 8 WMMAs
    int m  = m0 + (lane & 15);
    int kb = (lane >> 4) * 8;
    const _Float16* wrow = Wh + (size_t)m * C_ + kb;
    const _Float16* brow = sB + lane * 16;
    v8f acc[8] = {};                           // [t*2 + nt]
    #pragma unroll
    for (int kt = 0; kt < KT_; ++kt) {
        // A fragment (16x32 f16): lanes 0-15: M=lane, K = {0..7, 16..23};
        // lanes 16-31: M=lane-16, K = {8..15, 24..31}. Two b128 loads.
        AFrag af;
        af.h[0] = *(const v8h*)(wrow + kt * 32);
        af.h[1] = *(const v8h*)(wrow + kt * 32 + 16);
        #pragma unroll
        for (int t = 0; t < T_; ++t) {
            v16h bf0 = *(const v16h*)(brow + ((t * KT_ + kt) * 2 + 0) * 512);
            v16h bf1 = *(const v16h*)(brow + ((t * KT_ + kt) * 2 + 1) * 512);
            acc[t * 2 + 0] = __builtin_amdgcn_wmma_f32_16x16x32_f16(
                false, af.v, false, bf0, (short)0, acc[t * 2 + 0], false, false);
            acc[t * 2 + 1] = __builtin_amdgcn_wmma_f32_16x16x32_f16(
                false, af.v, false, bf1, (short)0, acc[t * 2 + 1], false, false);
        }
    }

    // ---- epilogue; C/D layout: VGPR r, lanes 0-15 -> M=r, lanes 16-31 -> M=r+8
    const float invsq = 1.f / sqrtf(1.f + 1e-5f);
    float wgt = (mode == 1) ? dense_w[b * E_ + expert] : 1.f;
    int mo = (lane < 16) ? 0 : 8;
    #pragma unroll
    for (int r = 0; r < 8; ++r) {
        int mm = m0 + r + mo;
        float ba = bias[mm], sc = bnS[mm] * invsq, bb = bnB[mm];
        #pragma unroll
        for (int nt = 0; nt < 2; ++nt) {
            int n = hw0 + nt * 16 + (lane & 15);
            size_t idx0 = ((size_t)b * C_ + mm) * HW_ + n;   // t stride = NSLICE
            if (mode == 0) {
                float v = 0.f;                                // fused LIF over t
                #pragma unroll
                for (int t = 0; t < T_; ++t) {
                    size_t idx = idx0 + (size_t)t * NSLICE;
                    float hv = (acc[t * 2 + nt][r] + ba) * sc + bb + resid[idx];
                    dst[idx] = hv;
                    v += (hv - v) * invTau;
                    float s = (v >= 1.f) ? 1.f : 0.f;
                    spkOut[idx] = (_Float16)s;
                    v = (s > 0.f) ? 0.f : v;
                }
            } else {
                #pragma unroll
                for (int t = 0; t < T_; ++t) {
                    size_t idx = idx0 + (size_t)t * NSLICE;
                    float vv = (acc[t * 2 + nt][r] + ba) * sc + bb + resid[idx];
                    dst[idx] += wgt * vv;
                }
            }
        }
    }
}

// ------------------------------------------------------------------- host
extern "C" void kernel_launch(void* const* d_in, const int* in_sizes, int n_in,
                              void* d_out, int out_size, void* d_ws, size_t ws_size,
                              hipStream_t stream) {
    const float* x   = (const float*)d_in[0];
    const float* rW  = (const float*)d_in[1];
    const float* rb  = (const float*)d_in[2];
    const float* rs  = (const float*)d_in[3];
    const float* rbb = (const float*)d_in[4];
    const float* W1  = (const float*)d_in[5];
    const float* b1  = (const float*)d_in[6];
    const float* s1  = (const float*)d_in[7];
    const float* bb1 = (const float*)d_in[8];
    const float* W2  = (const float*)d_in[9];
    const float* b2  = (const float*)d_in[10];
    const float* s2  = (const float*)d_in[11];
    const float* bb2 = (const float*)d_in[12];
    float* out = (float*)d_out;
    (void)in_sizes; (void)n_in; (void)ws_size;

    char* ws = (char*)d_ws;
    size_t off = 0;
    auto take = [&](size_t bytes) -> char* {
        char* p = ws + off;
        off += (bytes + 255) & ~(size_t)255;
        return p;
    };
    _Float16* W1h  = (_Float16*)take(sizeof(_Float16) * (size_t)E_ * C_ * C_);
    _Float16* W2h  = (_Float16*)take(sizeof(_Float16) * (size_t)E_ * C_ * C_);
    _Float16* spk1 = (_Float16*)take(sizeof(_Float16) * (size_t)T_ * NSLICE);
    _Float16* spk2 = (_Float16*)take(sizeof(_Float16) * (size_t)T_ * NSLICE);
    float*    hbuf = (float*)   take(sizeof(float)    * (size_t)T_ * NSLICE);
    float*    part = (float*)   take(sizeof(float)    * B_ * 8 * E_);
    float*    dense= (float*)   take(sizeof(float)    * B_ * E_);

    // router (deterministic two-stage reduction)
    router_partial<<<dim3(B_, 8), 256, 0, stream>>>(x, rW, part);
    router_topk<<<1, 32, 0, stream>>>(part, rb, rs, rbb, dense);

    // one-shot f32->f16 weight conversion for all experts
    int nw = E_ * C_ * C_;
    cvt_f32_f16<<<(nw + 255) / 256, 256, 0, stream>>>(W1, W1h, nw);
    cvt_f32_f16<<<(nw + 255) / 256, 256, 0, stream>>>(W2, W2h, nw);

    // output accumulator starts at zero
    zero_f32<<<(out_size + 255) / 256, 256, 0, stream>>>(out, out_size);

    for (int e = 0; e < E_; ++e) {
        float tau = 1.5f + (float)e * (2.5f / 7.f);   // linspace(1.5, 4.0, 8)
        float invTau = 1.f / tau;
        // LIF(x) -> stage-1 spikes
        lif_spike<<<NSLICE / 256, 256, 0, stream>>>(x, spk1, invTau);
        // GEMM1: h = x + BN(spk1 @ W1 + b1); epilogue also runs LIF(h) -> spk2
        expert_gemm_t4<<<dim3(128, 3), dim3(256), GEMM_LDS, stream>>>(
            spk1, W1h + (size_t)e * C_ * C_,
            b1 + e * C_, s1 + e * C_, bb1 + e * C_,
            x, hbuf, spk2, dense, invTau, e, /*mode=*/0);
        // GEMM2: out += gate * (BN(spk2 @ W2 + b2) + h)
        expert_gemm_t4<<<dim3(128, 3), dim3(256), GEMM_LDS, stream>>>(
            spk2, W2h + (size_t)e * C_ * C_,
            b2 + e * C_, s2 + e * C_, bb2 + e * C_,
            hbuf, out, spk1 /*unused*/, dense, invTau, e, /*mode=*/1);
    }
}
